// LSTMAutoencoder_30081950941189
// MI455X (gfx1250) — compile-verified, matrix operand order checked by reading, not run on hardware
//
#include <hip/hip_runtime.h>

// ---------------------------------------------------------------------------
// LSTM autoencoder for MI455X (gfx1250).  bf16 WMMA with f32 accumulate.
//   B=256, T=512, I=64, H=256, 4H=1024.
// ---------------------------------------------------------------------------

#define Bsz 256
#define Tsz 512
#define Isz 64
#define Hsz 256
#define N4  1024
#define NWG_REC 128   // 16 col-blocks x 8 batch-blocks, persistent

typedef __attribute__((ext_vector_type(16))) __bf16 v16bf;
typedef __attribute__((ext_vector_type(8)))  float  v8f;

union Frag { v16bf v; uint4 u[2]; };

__device__ __forceinline__ unsigned short f2bf(float f) {
    union { float f; unsigned int u; } v; v.f = f;
    unsigned int r = (v.u + 0x7FFFu + ((v.u >> 16) & 1u)) >> 16;  // RNE
    return (unsigned short)r;
}
__device__ __forceinline__ float bf2f(unsigned short h) {
    union { unsigned int u; float f; } v; v.u = ((unsigned int)h) << 16;
    return v.f;
}
__device__ __forceinline__ float sigm(float x) { return 1.f / (1.f + __expf(-x)); }

// ---------------------------------------------------------------------------
// init: zero the per-layer grid-barrier counters (fresh every launch)
// ---------------------------------------------------------------------------
__global__ void init_kernel(unsigned int* barCnt) {
    if (threadIdx.x < 8) barCnt[threadIdx.x] = 0u;
}

// ---------------------------------------------------------------------------
// elementwise f32 -> bf16
// ---------------------------------------------------------------------------
__global__ void conv_bf16_kernel(const float* __restrict__ src,
                                 unsigned short* __restrict__ dst, int n) {
    for (int i = blockIdx.x * blockDim.x + threadIdx.x; i < n;
         i += gridDim.x * blockDim.x)
        dst[i] = f2bf(src[i]);
}

// bias = bih + bhh
__global__ void bias_kernel(const float* __restrict__ a, const float* __restrict__ b,
                            float* __restrict__ o, int n) {
    for (int i = blockIdx.x * blockDim.x + threadIdx.x; i < n;
         i += gridDim.x * blockDim.x)
        o[i] = a[i] + b[i];
}

// ---------------------------------------------------------------------------
// Pack W[N][K] (row-major, f32) into B-fragment layout for wmma bf16 16x16x32.
// B[k][n] = W[n][k].  Per (ktile,ntile,lane): 16 bf16, lane holds column
// n = nt*16 + (lane&15), K chunk kb = kt*32 + (lane>>4)*8 : {kb..kb+7, kb+16..kb+23}.
// ---------------------------------------------------------------------------
__global__ void packB_kernel(const float* __restrict__ W,
                             unsigned short* __restrict__ dst, int K, int N) {
    int ntiles = N >> 4, ktiles = K >> 5;
    int total = ktiles * ntiles * 32;
    for (int idx = blockIdx.x * blockDim.x + threadIdx.x; idx < total;
         idx += gridDim.x * blockDim.x) {
        int lane = idx & 31, tile = idx >> 5;
        int nt = tile % ntiles, kt = tile / ntiles;
        int n  = nt * 16 + (lane & 15);
        int kb = kt * 32 + (lane >> 4) * 8;
        unsigned short* d = dst + (size_t)idx * 16;
        #pragma unroll
        for (int j = 0; j < 8; ++j) {
            d[j]     = f2bf(W[(size_t)n * K + kb + j]);
            d[8 + j] = f2bf(W[(size_t)n * K + kb + 16 + j]);
        }
    }
}

// ---------------------------------------------------------------------------
// Generic WMMA GEMM:  out[M,N] = A[M,K](bf16) * Bpacked + bias
// Each wave computes a 16x64 strip (4 accumulators), grid-stride over jobs.
// permuteBT: store row m (= b*T+t) at row (t*B + b)  [used for xp layout].
// ---------------------------------------------------------------------------
__global__ __launch_bounds__(256)
void wmma_gemm_kernel(const unsigned short* __restrict__ A,
                      const unsigned short* __restrict__ Bp,
                      const float* __restrict__ bias,
                      int M, int N, int K,
                      unsigned short* __restrict__ outB,
                      float* __restrict__ outF,
                      int permuteBT) {
    const int lane = threadIdx.x & 31;
    const int wid  = blockIdx.x * (blockDim.x >> 5) + (threadIdx.x >> 5);
    const int totW = gridDim.x * (blockDim.x >> 5);
    const int nGroups = N >> 6;              // groups of 64 columns
    const int jobs = (M >> 4) * nGroups;
    const int ntiles = N >> 4;
    const int kb0 = (lane >> 4) * 8;
    const int n0 = lane & 15, hi = lane >> 4;

    for (int job = wid; job < jobs; job += totW) {
        int mt = job / nGroups, ng = job % nGroups;
        int mBase = mt << 4, nBase = ng << 6;
        v8f acc0 = {}, acc1 = {}, acc2 = {}, acc3 = {};
        int row = mBase + n0;
        const uint4* Arow = (const uint4*)(A + (size_t)row * K);

        // prefetch next job's A rows into cache while this job computes
        {
            int nj = job + totW;
            if (nj < jobs) {
                int nrow = (nj / nGroups) * 16 + n0;
                __builtin_prefetch(A + (size_t)nrow * K + kb0, 0, 0);
            }
        }

        for (int kt = 0; kt < (K >> 5); ++kt) {
            Frag a;
            int ki = (kt * 32 + kb0) >> 3;   // uint4 index
            a.u[0] = Arow[ki];
            a.u[1] = Arow[ki + 2];
            const uint4* bp = (const uint4*)(Bp +
                ((size_t)(kt * ntiles + (nBase >> 4)) * 32 + lane) * 16);
            Frag b0, b1, b2, b3;
            b0.u[0] = bp[0];       b0.u[1] = bp[1];
            b1.u[0] = bp[0 + 64];  b1.u[1] = bp[1 + 64];   // next ntile = +32 lanes*16 = 64 uint4
            b2.u[0] = bp[0 + 128]; b2.u[1] = bp[1 + 128];
            b3.u[0] = bp[0 + 192]; b3.u[1] = bp[1 + 192];
            acc0 = __builtin_amdgcn_wmma_f32_16x16x32_bf16(false, a.v, false, b0.v, (short)0, acc0, false, false);
            acc1 = __builtin_amdgcn_wmma_f32_16x16x32_bf16(false, a.v, false, b1.v, (short)0, acc1, false, false);
            acc2 = __builtin_amdgcn_wmma_f32_16x16x32_bf16(false, a.v, false, b2.v, (short)0, acc2, false, false);
            acc3 = __builtin_amdgcn_wmma_f32_16x16x32_bf16(false, a.v, false, b3.v, (short)0, acc3, false, false);
        }
        v8f accs[4] = {acc0, acc1, acc2, acc3};
        #pragma unroll
        for (int j = 0; j < 4; ++j) {
            int n = nBase + j * 16 + n0;
            float bv = bias ? bias[n] : 0.f;
            #pragma unroll
            for (int r = 0; r < 8; ++r) {
                int m = mBase + r + 8 * hi;
                float v = accs[j][r] + bv;
                size_t orow = permuteBT ? ((size_t)(m & (Tsz - 1)) * Bsz + (m >> 9))
                                        : (size_t)m;
                if (outB) outB[orow * N + n] = f2bf(v);
                else      outF[orow * N + n] = v;
            }
        }
    }
}

// ---------------------------------------------------------------------------
// Persistent LSTM recurrence for one layer.
//   grid = 128 WGs: wgJ = wg&15 (16 h-columns), wgM = wg>>4 (32 batch rows).
//   Each WG computes all 4 gates for its columns -> local c/h update.
//   Whh slice lives in REGISTERS (8 fragments = 64 VGPRs per wave) for all
//   512 steps; c state lives in LDS.  Device-scope spin barrier per step;
//   h ping-pongs via global (L2-resident: 128 KB vs 192 MB L2).
// ---------------------------------------------------------------------------
__global__ __launch_bounds__(256)
void lstm_rec_kernel(const unsigned short* __restrict__ xp,   // [T*B, 4H], row = t*B+b
                     const unsigned short* __restrict__ WhhP, // packed, ntiles=64
                     unsigned short* __restrict__ hseq,       // [B*T, H], row = b*T+t
                     unsigned short* __restrict__ hbuf,       // [2, B, H]
                     unsigned int* __restrict__ barCnt) {
    __shared__ float sGate[2][4][16][16];                          // 8 KB
    __shared__ float sC[32][16];                                   // 2 KB

    const int tid = threadIdx.x, lane = tid & 31, wv = tid >> 5;   // 8 waves
    const int wg = blockIdx.x;
    const int wgJ = wg & 15, wgM = wg >> 4;
    const int jBase = wgJ * 16, mBase = wgM * 32;
    const int mt = wv & 1, gate = wv >> 1;
    const int n0 = lane & 15, hi = lane >> 4;
    const int kb0 = hi * 8;

    // register-stationary Whh fragments for this wave's (gate, column-block)
    Frag bfr[8];
    {
        int nt = gate * 16 + wgJ;                      // ntiles = 64
        #pragma unroll
        for (int kt = 0; kt < 8; ++kt) {
            const uint4* bp = (const uint4*)(WhhP +
                ((size_t)(kt * 64 + nt) * 32 + lane) * 16);
            bfr[kt].u[0] = bp[0];
            bfr[kt].u[1] = bp[1];
        }
    }
    for (int e = tid; e < 32 * 16; e += blockDim.x) sC[e >> 4][e & 15] = 0.f;
    __syncthreads();

    const int rowA = mBase + mt * 16 + n0;             // batch row this lane loads
    const int nCol = gate * Hsz + jBase + n0;          // xp column this lane reads

    for (int t = 0; t < Tsz; ++t) {
        // prefetch next step's xp rows (this wave reads 8 rows x 1 col each)
        if (t + 1 < Tsz) {
            int m0 = mBase + mt * 16 + 8 * hi;
            __builtin_prefetch(xp + ((size_t)(t + 1) * Bsz + m0) * N4 + nCol, 0, 0);
        }

        v8f acc = {};
        if (t > 0) {
            const unsigned short* hprev = hbuf + (size_t)(t & 1) * Bsz * Hsz;
            const uint4* Arow = (const uint4*)(hprev + (size_t)rowA * Hsz);
            #pragma unroll
            for (int kt = 0; kt < 8; ++kt) {
                Frag a;
                int ki = (kt * 32 + kb0) >> 3;
                a.u[0] = Arow[ki];
                a.u[1] = Arow[ki + 2];
                acc = __builtin_amdgcn_wmma_f32_16x16x32_bf16(false, a.v, false, bfr[kt].v, (short)0, acc, false, false);
            }
        }
        // add xp (pre-gates from x side) and stash tile to LDS
        #pragma unroll
        for (int r = 0; r < 8; ++r) {
            int m = mBase + mt * 16 + r + 8 * hi;      // batch row
            float xv = bf2f(xp[((size_t)t * Bsz + m) * N4 + nCol]);
            sGate[mt][gate][r + 8 * hi][n0] = acc[r] + xv;
        }
        __syncthreads();

        // gate combine + cell update: 512 elems, 256 threads
        unsigned short* hnext = hbuf + (size_t)((t + 1) & 1) * Bsz * Hsz;
        for (int e = tid; e < 32 * 16; e += blockDim.x) {
            int rrow = e >> 4, col = e & 15;
            int mtt = rrow >> 4, mr = rrow & 15;
            float i = sigm(sGate[mtt][0][mr][col]);
            float f = sigm(sGate[mtt][1][mr][col]);
            float g = tanhf(sGate[mtt][2][mr][col]);
            float o = sigm(sGate[mtt][3][mr][col]);
            float c = f * sC[rrow][col] + i * g;
            sC[rrow][col] = c;
            float h = o * tanhf(c);
            unsigned short hb = f2bf(h);
            int b = mBase + rrow;
            hnext[(size_t)b * Hsz + jBase + col] = hb;
            hseq[((size_t)b * Tsz + t) * Hsz + jBase + col] = hb;
        }
        __threadfence();
        __syncthreads();

        // device-scope barrier: everyone's h for step t must be visible
        if (tid == 0) {
            __hip_atomic_fetch_add(barCnt, 1u, __ATOMIC_ACQ_REL, __HIP_MEMORY_SCOPE_AGENT);
            unsigned int target = (unsigned int)(t + 1) * NWG_REC;
            while (__hip_atomic_load(barCnt, __ATOMIC_ACQUIRE, __HIP_MEMORY_SCOPE_AGENT) < target)
                __builtin_amdgcn_s_sleep(1);
        }
        __syncthreads();
    }
}

// ---------------------------------------------------------------------------
// Host orchestration
// ---------------------------------------------------------------------------
extern "C" void kernel_launch(void* const* d_in, const int* in_sizes, int n_in,
                              void* d_out, int out_size, void* d_ws, size_t ws_size,
                              hipStream_t stream) {
    (void)in_sizes; (void)n_in; (void)out_size; (void)ws_size;
    const float* x    = (const float*)d_in[0];
    const float* outW = (const float*)d_in[17];
    const float* outb = (const float*)d_in[18];

    // carve workspace
    size_t off = 0;
    auto carve = [&](size_t bytes) -> char* {
        off = (off + 255) & ~(size_t)255;
        char* p = (char*)d_ws + off;
        off += bytes;
        return p;
    };
    unsigned int*   barCnt = (unsigned int*)carve(8 * sizeof(unsigned int));
    unsigned short* xbf    = (unsigned short*)carve((size_t)Bsz * Tsz * Isz * 2);
    unsigned short* hseqA  = (unsigned short*)carve((size_t)Bsz * Tsz * Hsz * 2);
    unsigned short* hseqB  = (unsigned short*)carve((size_t)Bsz * Tsz * Hsz * 2);
    unsigned short* xpbuf  = (unsigned short*)carve((size_t)Bsz * Tsz * N4 * 2);
    unsigned short* hbuf   = (unsigned short*)carve((size_t)2 * Bsz * Hsz * 2);
    unsigned short* WihP[4], * WhhP[4];
    float* biasC[4];
    for (int l = 0; l < 4; ++l) {
        WihP[l]  = (unsigned short*)carve((size_t)256 * 1024 * 2);  // max size
        WhhP[l]  = (unsigned short*)carve((size_t)256 * 1024 * 2);
        biasC[l] = (float*)carve(N4 * sizeof(float));
    }
    unsigned short* outWP = (unsigned short*)carve((size_t)256 * 64 * 2);

    init_kernel<<<1, 64, 0, stream>>>(barCnt);
    conv_bf16_kernel<<<512, 256, 0, stream>>>(x, xbf, Bsz * Tsz * Isz);

    for (int l = 0; l < 4; ++l) {
        const float* Wih = (const float*)d_in[1 + 4 * l];
        const float* Whh = (const float*)d_in[2 + 4 * l];
        const float* bih = (const float*)d_in[3 + 4 * l];
        const float* bhh = (const float*)d_in[4 + 4 * l];
        int Kin = (l == 0) ? Isz : Hsz;
        packB_kernel<<<256, 256, 0, stream>>>(Wih, WihP[l], Kin, N4);
        packB_kernel<<<256, 256, 0, stream>>>(Whh, WhhP[l], Hsz, N4);
        bias_kernel<<<4, 256, 0, stream>>>(bih, bhh, biasC[l], N4);
    }
    packB_kernel<<<64, 256, 0, stream>>>(outW, outWP, Hsz, Isz);

    const int M = Bsz * Tsz;
    const unsigned short* cur = xbf;
    unsigned short* hOut = hseqA;
    unsigned short* hAlt = hseqB;
    int Kcur = Isz;
    for (int l = 0; l < 4; ++l) {
        // xp[t*B+b, 4H] = cur @ Wih^T + (bih+bhh)
        wmma_gemm_kernel<<<2048, 256, 0, stream>>>(
            cur, WihP[l], biasC[l], M, N4, Kcur, xpbuf, nullptr, 1);
        // recurrent sweep
        lstm_rec_kernel<<<NWG_REC, 256, 0, stream>>>(
            xpbuf, WhhP[l], hOut, hbuf, barCnt + l);
        cur = hOut;
        unsigned short* tmp = hOut; hOut = hAlt; hAlt = tmp;
        Kcur = Hsz;
    }

    // reconstruction[b*T+t, I] = hseq @ outW^T + out_b   (f32 out)
    wmma_gemm_kernel<<<1024, 256, 0, stream>>>(
        cur, outWP, outb, M, Isz, Hsz, nullptr, (float*)d_out, 0);
}
